// CompositionalPID_57406532878594
// MI455X (gfx1250) — matrix operand order repaired and checked by reading, not compiled, not run on hardware
//
#include <hip/hip_runtime.h>
#include <hip/hip_bf16.h>
#include <math.h>

#define DEV __device__ __forceinline__

typedef __attribute__((ext_vector_type(16))) __bf16 v16bf;
typedef __attribute__((ext_vector_type(8)))  __bf16 v8bf;
typedef __attribute__((ext_vector_type(8)))  float  v8f;

namespace {

constexpr int Bb = 16, Ll = 64, Dd = 768, FFf = 2048, Hh = 8, HDh = 96;
constexpr int NCc = 8, Nn = 256;
constexpr int BL  = Bb * Ll;        // 1024 tokens
constexpr int TOT = Bb * NCc * Nn;  // 32768 rows for operator GEMMs
constexpr int BN  = Bb * Nn;        // 4096 rows for composer GEMMs

// ---------- fragment loads (bf16 in memory; no per-element ALU) ----------
// A fragment 16x32 (MxK): lane l -> m=l%16, half=l/16.
// memory: A[m, k0+half*16 .. +15] (16 contiguous K values)
DEV v16bf load_a_frag(const __bf16* ap) {
  return *(const v16bf*)(ap);
}
// B fragment 32x16 (KxN) from W[N,K] row-major: lane l -> n=l%16, half=l/16;
// holds 16 contiguous K values starting at k0 + half*16 (32B aligned).
DEV v16bf load_b_frag(const __bf16* wp) {
  return *(const v16bf*)(wp);
}
DEV v8f wmma_bf16(v16bf a, v16bf b, v8f c) {
  return __builtin_amdgcn_wmma_f32_16x16x32_bf16(false, a, false, b, (short)0, c, false, false);
}

// ---------- WMMA GEMM: C[M,N] = A1 @ W[:, :K]^T (+ A2 @ W[:, K:2K]^T) + bias ----------
// A, W in bf16; C fp32 (+ optional bf16 copy).
// TWO=1: wave computes 32(M) x 64(N); TWO=0: 16(M) x 64(N) (for M==16).
// Software-pipelined: next K-step fragments are loaded before current WMMAs.
template<int RELU, int TWO>
__global__ __launch_bounds__(256) void k_gemm(
    const __bf16* __restrict__ A1, const __bf16* __restrict__ A2, int lda,
    const __bf16* __restrict__ W, int ldw, const float* __restrict__ bias,
    float* __restrict__ C, __bf16* __restrict__ Cbf, int M, int N, int K)
{
  const int ntiles = N >> 6;
  const int msh    = TWO ? 5 : 4;
  const int mtiles = M >> msh;
  const int total  = mtiles * ntiles;
  const int wid  = threadIdx.x >> 5;
  const int lane = threadIdx.x & 31;
  const int t = blockIdx.x * 8 + wid;     // uniform per wave
  if (t >= total) return;                 // wave-uniform: EXEC stays all-ones
  const int mtile = t / ntiles;
  const int ntile = t - mtile * ntiles;
  const int half  = lane >> 4;
  const int l16   = lane & 15;
  const int mbase = mtile << msh;

  const __bf16* wpb = W + (size_t)((ntile << 6) + l16) * ldw + half * 16;
  const size_t wstep = (size_t)16 * ldw;

  v8f acc[2][4];
  #pragma unroll
  for (int s = 0; s < 2; ++s)
    #pragma unroll
    for (int f = 0; f < 4; ++f)
      acc[s][f] = v8f{0.f,0.f,0.f,0.f,0.f,0.f,0.f,0.f};

  const int nph = (A2 != nullptr) ? 2 : 1;
  for (int ph = 0; ph < nph; ++ph) {
    const __bf16* A = ph ? A2 : A1;
    const __bf16* ap0 = A + (size_t)(mbase + l16) * lda + half * 16;
    const __bf16* ap1 = ap0 + (size_t)16 * lda;
    const __bf16* w0 = wpb + ph * K;
    const __bf16* w1 = w0 + wstep;
    const __bf16* w2 = w1 + wstep;
    const __bf16* w3 = w2 + wstep;

    // prologue loads
    v16bf a0 = load_a_frag(ap0);
    v16bf a1 = a0;
    if (TWO) a1 = load_a_frag(ap1);
    v16bf b0 = load_b_frag(w0);
    v16bf b1 = load_b_frag(w1);
    v16bf b2 = load_b_frag(w2);
    v16bf b3 = load_b_frag(w3);

    #pragma unroll 2
    for (int k0 = 32; k0 < K; k0 += 32) {
      // issue next-step loads first so they overlap the WMMAs below
      v16bf a0n = load_a_frag(ap0 + k0);
      v16bf a1n = a0n;
      if (TWO) a1n = load_a_frag(ap1 + k0);
      v16bf b0n = load_b_frag(w0 + k0);
      v16bf b1n = load_b_frag(w1 + k0);
      v16bf b2n = load_b_frag(w2 + k0);
      v16bf b3n = load_b_frag(w3 + k0);

      acc[0][0] = wmma_bf16(a0, b0, acc[0][0]);
      acc[0][1] = wmma_bf16(a0, b1, acc[0][1]);
      acc[0][2] = wmma_bf16(a0, b2, acc[0][2]);
      acc[0][3] = wmma_bf16(a0, b3, acc[0][3]);
      if (TWO) {
        acc[1][0] = wmma_bf16(a1, b0, acc[1][0]);
        acc[1][1] = wmma_bf16(a1, b1, acc[1][1]);
        acc[1][2] = wmma_bf16(a1, b2, acc[1][2]);
        acc[1][3] = wmma_bf16(a1, b3, acc[1][3]);
      }
      a0 = a0n; a1 = a1n;
      b0 = b0n; b1 = b1n; b2 = b2n; b3 = b3n;
    }
    // pipeline epilogue
    acc[0][0] = wmma_bf16(a0, b0, acc[0][0]);
    acc[0][1] = wmma_bf16(a0, b1, acc[0][1]);
    acc[0][2] = wmma_bf16(a0, b2, acc[0][2]);
    acc[0][3] = wmma_bf16(a0, b3, acc[0][3]);
    if (TWO) {
      acc[1][0] = wmma_bf16(a1, b0, acc[1][0]);
      acc[1][1] = wmma_bf16(a1, b1, acc[1][1]);
      acc[1][2] = wmma_bf16(a1, b2, acc[1][2]);
      acc[1][3] = wmma_bf16(a1, b3, acc[1][3]);
    }
  }

  #pragma unroll
  for (int f = 0; f < 4; ++f) {
    int n = (ntile << 6) + f * 16 + l16;
    float bb = bias ? bias[n] : 0.f;
    #pragma unroll
    for (int r = 0; r < 8; ++r) {
      int m = mbase + half * 8 + r;
      float v = acc[0][f][r] + bb;
      if (RELU) v = fmaxf(v, 0.f);
      C[(size_t)m * N + n] = v;
      if (Cbf) Cbf[(size_t)m * N + n] = (__bf16)v;
    }
    if (TWO) {
      #pragma unroll
      for (int r = 0; r < 8; ++r) {
        int m = mbase + 16 + half * 8 + r;
        float v = acc[1][f][r] + bb;
        if (RELU) v = fmaxf(v, 0.f);
        C[(size_t)m * N + n] = v;
        if (Cbf) Cbf[(size_t)m * N + n] = (__bf16)v;
      }
    }
  }
}

// ---------- fp32 -> bf16 bulk convert ----------
__global__ __launch_bounds__(256) void k_cvt(
    const float* __restrict__ in, __bf16* __restrict__ out, long n)
{
  long i = (long)blockIdx.x * 256 + threadIdx.x;
  long stride = (long)gridDim.x * 256;
  for (; i < n; i += stride) out[i] = (__bf16)in[i];
}

// ---------- attention: one block per (b, h); writes bf16 output ----------
__global__ __launch_bounds__(256) void k_attn(
    const float* __restrict__ qkv, const unsigned char* __restrict__ mask,
    __bf16* __restrict__ attnbf)
{
  __shared__ float sKV[Ll * HDh];  // K, later reused for V
  __shared__ float sS[Ll * Ll];    // scores/probs
  const int b = blockIdx.x / Hh;
  const int h = blockIdx.x % Hh;
  const int tid = threadIdx.x;
  const float scale = 0.10206207261596575f;  // 1/sqrt(96)

  for (int i = tid; i < Ll * HDh; i += 256) {
    int m = i / HDh, d2 = i - m * HDh;
    sKV[i] = qkv[(size_t)(b * Ll + m) * (3 * Dd) + Dd + h * HDh + d2];
  }
  __syncthreads();
  for (int i = tid; i < Ll * Ll; i += 256) {
    int l = i >> 6, m = i & 63;
    const float* qp = qkv + (size_t)(b * Ll + l) * (3 * Dd) + h * HDh;
    const float* kp = sKV + m * HDh;
    float s = 0.f;
    for (int d2 = 0; d2 < HDh; ++d2) s += qp[d2] * kp[d2];
    s *= scale;
    if (!mask[b * Ll + m]) s = -1e9f;
    sS[i] = s;
  }
  __syncthreads();
  if (tid < Ll) {
    float mx = -2e9f;
    for (int m = 0; m < Ll; ++m) mx = fmaxf(mx, sS[tid * Ll + m]);
    float sum = 0.f;
    for (int m = 0; m < Ll; ++m) sum += __expf(sS[tid * Ll + m] - mx);
    float inv = 1.f / sum;
    for (int m = 0; m < Ll; ++m) sS[tid * Ll + m] = __expf(sS[tid * Ll + m] - mx) * inv;
  }
  __syncthreads();
  for (int i = tid; i < Ll * HDh; i += 256) {
    int m = i / HDh, d2 = i - m * HDh;
    sKV[i] = qkv[(size_t)(b * Ll + m) * (3 * Dd) + 2 * Dd + h * HDh + d2];
  }
  __syncthreads();
  for (int i = tid; i < Ll * HDh; i += 256) {
    int l = i / HDh, d2 = i - l * HDh;
    float o = 0.f;
    for (int m = 0; m < Ll; ++m) o += sS[l * Ll + m] * sKV[m * HDh + d2];
    attnbf[(size_t)(b * Ll + l) * Dd + h * HDh + d2] = (__bf16)o;
  }
}

// ---------- LayerNorm (optional residual / relu); fp32 out + optional bf16 out ----------
template<int RES, int RELU>
__global__ __launch_bounds__(256) void k_add_ln(
    const float* __restrict__ in, const float* __restrict__ res,
    const float* __restrict__ w, const float* __restrict__ b,
    float* __restrict__ out, __bf16* __restrict__ outbf, int Dlen)
{
  __shared__ float red[256];
  const int row = blockIdx.x, tid = threadIdx.x;
  const float* ip = in + (size_t)row * Dlen;
  const float* rp = RES ? (res + (size_t)row * Dlen) : nullptr;
  float* op = out + (size_t)row * Dlen;
  __bf16* ob = outbf ? (outbf + (size_t)row * Dlen) : nullptr;
  const int nper = Dlen >> 8;  // 3 (768) or 6 (1536)
  float v[6];
  float s = 0.f;
  for (int i = 0; i < nper; ++i) {
    int c = i * 256 + tid;
    float x = ip[c];
    if (RES) x += rp[c];
    v[i] = x; s += x;
  }
  red[tid] = s; __syncthreads();
  for (int st = 128; st > 0; st >>= 1) { if (tid < st) red[tid] += red[tid + st]; __syncthreads(); }
  float mean = red[0] / (float)Dlen; __syncthreads();
  float s2 = 0.f;
  for (int i = 0; i < nper; ++i) { float d2 = v[i] - mean; s2 += d2 * d2; }
  red[tid] = s2; __syncthreads();
  for (int st = 128; st > 0; st >>= 1) { if (tid < st) red[tid] += red[tid + st]; __syncthreads(); }
  float rstd = rsqrtf(red[0] / (float)Dlen + 1e-5f);
  for (int i = 0; i < nper; ++i) {
    int c = i * 256 + tid;
    float y = (v[i] - mean) * rstd * w[c] + b[c];
    if (RELU) y = y > 0.f ? y : 0.f;
    op[c] = y;
    if (ob) ob[c] = (__bf16)y;
  }
}

// ---------- operator epilogue: operated(+)= sel[b,i,k] * LN(o_row) ----------
template<int INIT>
__global__ __launch_bounds__(256) void k_op_ln_axpy(
    const float* __restrict__ o, const float* __restrict__ w, const float* __restrict__ b,
    const float* __restrict__ sel, int kk, float* __restrict__ operated)
{
  __shared__ float red[256];
  const int row = blockIdx.x, tid = threadIdx.x;
  const float* ip = o + (size_t)row * Dd;
  float* op = operated + (size_t)row * Dd;
  const float sv = sel[(row >> 8) * 4 + kk];  // row -> (b*8+i) = row/256
  float v[3];
  float s = 0.f;
  for (int i = 0; i < 3; ++i) { int c = i * 256 + tid; v[i] = ip[c]; s += v[i]; }
  red[tid] = s; __syncthreads();
  for (int st = 128; st > 0; st >>= 1) { if (tid < st) red[tid] += red[tid + st]; __syncthreads(); }
  float mean = red[0] / (float)Dd; __syncthreads();
  float s2 = 0.f;
  for (int i = 0; i < 3; ++i) { float d2 = v[i] - mean; s2 += d2 * d2; }
  red[tid] = s2; __syncthreads();
  for (int st = 128; st > 0; st >>= 1) { if (tid < st) red[tid] += red[tid + st]; __syncthreads(); }
  float rstd = rsqrtf(red[0] / (float)Dd + 1e-5f);
  for (int i = 0; i < 3; ++i) {
    int c = i * 256 + tid;
    float y = (v[i] - mean) * rstd * w[c] + b[c];
    if (INIT) op[c] = sv * y; else op[c] += sv * y;
  }
}

// ---------- mean over R rows of [R, Dlen] slab; optional bf16 dual output ----------
__global__ __launch_bounds__(256) void k_mean_rows(
    const float* __restrict__ in, float* __restrict__ out,
    __bf16* __restrict__ outbf, int R, int Dlen)
{
  const int sl = blockIdx.x;
  const float* ip = in + (size_t)sl * R * Dlen;
  float* op = out + (size_t)sl * Dlen;
  __bf16* ob = outbf ? (outbf + (size_t)sl * Dlen) : nullptr;
  const float inv = 1.f / (float)R;
  for (int c = threadIdx.x; c < Dlen; c += 256) {
    float acc = 0.f;
    for (int r = 0; r < R; ++r) acc += ip[(size_t)r * Dlen + c];
    float y = acc * inv;
    op[c] = y;
    if (ob) ob[c] = (__bf16)y;
  }
}

// ---------- softmax over last dim of [B*8, 8] in-place ----------
__global__ void k_softmax8(float* __restrict__ cm)
{
  int t = threadIdx.x;
  if (t >= Bb * NCc) return;
  float* r = cm + t * 8;
  float m = r[0];
  for (int j = 1; j < 8; ++j) m = fmaxf(m, r[j]);
  float e[8], s = 0.f;
  for (int j = 0; j < 8; ++j) { e[j] = __expf(r[j] - m); s += e[j]; }
  float inv = 1.f / s;
  for (int j = 0; j < 8; ++j) r[j] = e[j] * inv;
}

// ---------- combined[b,i,n,:] = sum_j cw[b,i,j] * components[b,n,j,:] (fp32 + bf16) ----------
__global__ __launch_bounds__(256) void k_combined(
    const float* __restrict__ comps, const float* __restrict__ cw,
    float* __restrict__ combined, __bf16* __restrict__ combbf)
{
  __shared__ float sc[NCc * Dd];  // 24KB
  __shared__ float scw[64];
  const int bn = blockIdx.x;
  const int b = bn >> 8, n = bn & 255;
  const int tid = threadIdx.x;
  const float* base = comps + (size_t)(b * Nn + n) * NCc * Dd;
  for (int i = tid; i < NCc * Dd; i += 256) sc[i] = base[i];
  if (tid < 64) scw[tid] = cw[b * 64 + tid];
  __syncthreads();
  for (int i = 0; i < NCc; ++i) {
    for (int c = tid; c < Dd; c += 256) {
      float a = 0.f;
      #pragma unroll
      for (int j = 0; j < NCc; ++j) a += scw[i * 8 + j] * sc[j * Dd + c];
      size_t idx = (size_t)((b * NCc + i) * Nn + n) * Dd + c;
      combined[idx] = a;
      combbf[idx] = (__bf16)a;
    }
  }
}

// ---------- sel = softmax(Linear(cat(mean_n(combined), pooled))) ----------
__global__ __launch_bounds__(512) void k_sel(
    const float* __restrict__ selmean, const float* __restrict__ pooled,
    const float* __restrict__ sw, const float* __restrict__ sb,
    float* __restrict__ sel)
{
  __shared__ float lg[512];
  const int tid = threadIdx.x;
  const int rowi = tid >> 2;   // 0..127 = b*8+i
  const int kk = tid & 3;
  const int b = rowi >> 3;
  const float* a = selmean + (size_t)rowi * Dd;
  const float* p = pooled + (size_t)b * Dd;
  const float* w1 = sw + (size_t)kk * (2 * Dd);
  float s = sb[kk];
  for (int j = 0; j < Dd; ++j) s += a[j] * w1[j] + p[j] * w1[Dd + j];
  lg[tid] = s;
  __syncthreads();
  if (kk == 0) {
    float m = fmaxf(fmaxf(lg[tid], lg[tid + 1]), fmaxf(lg[tid + 2], lg[tid + 3]));
    float e0 = __expf(lg[tid] - m), e1 = __expf(lg[tid + 1] - m);
    float e2 = __expf(lg[tid + 2] - m), e3 = __expf(lg[tid + 3] - m);
    float inv = 1.f / (e0 + e1 + e2 + e3);
    sel[rowi * 4 + 0] = e0 * inv; sel[rowi * 4 + 1] = e1 * inv;
    sel[rowi * 4 + 2] = e2 * inv; sel[rowi * 4 + 3] = e3 * inv;
  }
}

// ---------- composer gate ----------
__global__ __launch_bounds__(256) void k_gate(
    const float* __restrict__ cmean, const float* __restrict__ cw,
    const float* __restrict__ cb, float* __restrict__ gate, int depth)
{
  __shared__ float red[256];
  const int tid = threadIdx.x;
  const int b = tid >> 4, seg = tid & 15;
  float s = 0.f;
  for (int j = seg * 48; j < seg * 48 + 48; ++j) s += cmean[(size_t)b * Dd + j] * cw[j];
  red[tid] = s;
  __syncthreads();
  if (seg == 0) {
    float t = cb[0];
    for (int i = 0; i < 16; ++i) t += red[tid + i];
    red[tid] = 1.f / (1.f + __expf(-t));
  }
  __syncthreads();
  if (tid == 0) {
    float p = 0.f;
    for (int b2 = 0; b2 < 16; ++b2) p += red[b2 * 16];
    p *= (1.f / 16.f);
    gate[0] = (depth == 0) ? 1.f : (gate[0] * (p >= 0.5f ? 1.f : 0.f));
  }
}

// ---------- gather feats[i] -> optional fp32 out + bf16 out ----------
__global__ __launch_bounds__(256) void k_gather(
    const float* __restrict__ operated, float* __restrict__ outf,
    __bf16* __restrict__ outb, int i)
{
  size_t idx = (size_t)blockIdx.x * 256 + threadIdx.x;
  int d = (int)(idx % Dd);
  size_t r = idx / Dd;
  int n = (int)(r & 255);
  int b = (int)(r >> 8);
  float v = operated[((size_t)(b * NCc + i) * Nn + n) * Dd + d];
  if (outf) outf[idx] = v;
  outb[idx] = (__bf16)v;
}

// ---------- cur = gate ? nb : cur; curbf = bf16(cur) ----------
__global__ __launch_bounds__(256) void k_blend(
    float* __restrict__ cur, const float* __restrict__ nb,
    const float* __restrict__ gate, __bf16* __restrict__ curbf)
{
  size_t idx = (size_t)blockIdx.x * 256 + threadIdx.x;
  float v = (gate[0] > 0.5f) ? nb[idx] : cur[idx];
  cur[idx] = v;
  curbf[idx] = (__bf16)v;
}

}  // namespace

extern "C" void kernel_launch(void* const* d_in, const int* in_sizes, int n_in,
                              void* d_out, int out_size, void* d_ws, size_t ws_size,
                              hipStream_t stream) {
  (void)in_sizes; (void)n_in; (void)out_size; (void)ws_size;
  const float* components = (const float*)d_in[0];
  const float* text_feats = (const float*)d_in[1];
  const unsigned char* mask = (const unsigned char*)d_in[2];
  auto P = [&](int i) { return (const float*)d_in[i]; };
  // Param flattening (depth-first, dict insertion order):
  // enc layer l: base 3+l*12: qkv_w,qkv_b,out_w,out_b,ln1_w,ln1_b,lin1_w,lin1_b,lin2_w,lin2_b,ln2_w,ln2_b
  // 39..42 opgen w1,b1,w2,b2 ; 43 sel_w ; 44 sel_b
  // operator k: base 45+k*8: w1,b1,ln1_w,ln1_b,w2,b2,ln2_w,ln2_b
  // 77 cont_w ; 78 cont_b ; composer layer d: base 79+d*4: w,b,ln_w,ln_b

  char* wsb = (char*)d_ws;
  size_t off = 0;
  auto allocF = [&](size_t n) { float* p = (float*)(wsb + off); off += ((n * 4 + 255) & ~(size_t)255); return p; };
  auto allocB = [&](size_t n) { __bf16* p = (__bf16*)(wsb + off); off += ((n * 2 + 255) & ~(size_t)255); return p; };

  // fp32 buffers
  float* qkvb      = allocF((size_t)BL * 3 * Dd);
  float* xbuf      = allocF((size_t)BL * Dd);
  float* projb     = allocF((size_t)BL * Dd);
  float* ffb       = allocF((size_t)BL * FFf);
  float* ff2b      = allocF((size_t)BL * Dd);
  float* pooled    = allocF((size_t)Bb * Dd);
  float* ogh       = allocF((size_t)Bb * 2 * Dd);
  float* cwb       = allocF((size_t)Bb * 64);
  float* selmean   = allocF((size_t)Bb * NCc * Dd);
  float* selb      = allocF((size_t)Bb * NCc * 4);
  float* cmean     = allocF((size_t)Bb * Dd);
  float* gate      = allocF(64);
  float* combinedb = allocF((size_t)TOT * Dd);
  float* oph       = allocF((size_t)TOT * 2 * Dd);
  float* opo       = allocF((size_t)TOT * Dd);
  float* operatedb = allocF((size_t)TOT * Dd);
  float* curb      = allocF((size_t)BN * Dd);
  float* nb0       = allocF((size_t)BN * Dd);
  float* nb1       = allocF((size_t)BN * Dd);

  // bf16 activation buffers
  __bf16* xbf      = allocB((size_t)BL * Dd);
  __bf16* attnbf   = allocB((size_t)BL * Dd);
  __bf16* ffbf     = allocB((size_t)BL * FFf);
  __bf16* pooledbf = allocB((size_t)Bb * Dd);
  __bf16* oghbf    = allocB((size_t)Bb * 2 * Dd);
  __bf16* combbf   = allocB((size_t)TOT * Dd);
  __bf16* ophbf    = allocB((size_t)TOT * 2 * Dd);
  __bf16* curbf    = allocB((size_t)BN * Dd);
  __bf16* nb0bf    = allocB((size_t)BN * Dd);
  __bf16* nb1bf    = allocB((size_t)BN * Dd);
  __bf16* fbufbf   = allocB((size_t)BN * Dd);

  // bf16 weight buffers
  __bf16 *wqkv[3], *wout[3], *wlin1[3], *wlin2[3], *wop1[4], *wop2[4], *wcomp[3];
  for (int l = 0; l < 3; ++l) {
    wqkv[l]  = allocB((size_t)3 * Dd * Dd);
    wout[l]  = allocB((size_t)Dd * Dd);
    wlin1[l] = allocB((size_t)FFf * Dd);
    wlin2[l] = allocB((size_t)Dd * FFf);
  }
  __bf16* wog1 = allocB((size_t)2 * Dd * Dd);
  __bf16* wog2 = allocB((size_t)64 * 2 * Dd);
  for (int k = 0; k < 4; ++k) {
    wop1[k] = allocB((size_t)2 * Dd * Dd);
    wop2[k] = allocB((size_t)Dd * 2 * Dd);
  }
  for (int d = 0; d < 3; ++d) wcomp[d] = allocB((size_t)Dd * 2 * Dd);

  auto cvt = [&](const float* in, __bf16* out, size_t n) {
    int blocks = (int)((n + 255) / 256); if (blocks > 2048) blocks = 2048;
    k_cvt<<<blocks, 256, 0, stream>>>(in, out, (long)n);
  };
  auto gemm = [&](const __bf16* A1, const __bf16* A2, int lda, const __bf16* W, int ldw,
                  const float* bias, float* C, __bf16* Cbf, int M, int N, int K, bool relu) {
    const bool two = (M % 32) == 0;
    int tiles = (two ? (M >> 5) : (M >> 4)) * (N >> 6);
    int blocks = (tiles + 7) >> 3;
    if (two) {
      if (relu) k_gemm<1,1><<<blocks, 256, 0, stream>>>(A1, A2, lda, W, ldw, bias, C, Cbf, M, N, K);
      else      k_gemm<0,1><<<blocks, 256, 0, stream>>>(A1, A2, lda, W, ldw, bias, C, Cbf, M, N, K);
    } else {
      if (relu) k_gemm<1,0><<<blocks, 256, 0, stream>>>(A1, A2, lda, W, ldw, bias, C, Cbf, M, N, K);
      else      k_gemm<0,0><<<blocks, 256, 0, stream>>>(A1, A2, lda, W, ldw, bias, C, Cbf, M, N, K);
    }
  };

  // ===== one-time (per launch) weight conversion to bf16 =====
  for (int l = 0; l < 3; ++l) {
    int e = 3 + l * 12;
    cvt(P(e + 0), wqkv[l],  (size_t)3 * Dd * Dd);
    cvt(P(e + 2), wout[l],  (size_t)Dd * Dd);
    cvt(P(e + 6), wlin1[l], (size_t)FFf * Dd);
    cvt(P(e + 8), wlin2[l], (size_t)Dd * FFf);
  }
  cvt(P(39), wog1, (size_t)2 * Dd * Dd);
  cvt(P(41), wog2, (size_t)64 * 2 * Dd);
  for (int k = 0; k < 4; ++k) {
    int o = 45 + k * 8;
    cvt(P(o + 0), wop1[k], (size_t)2 * Dd * Dd);
    cvt(P(o + 4), wop2[k], (size_t)Dd * 2 * Dd);
  }
  for (int d = 0; d < 3; ++d) cvt(P(79 + d * 4), wcomp[d], (size_t)Dd * 2 * Dd);

  // ===== encoder =====
  hipMemcpyAsync(xbuf, text_feats, (size_t)BL * Dd * sizeof(float),
                 hipMemcpyDeviceToDevice, stream);
  cvt(text_feats, xbf, (size_t)BL * Dd);
  for (int l = 0; l < 3; ++l) {
    int e = 3 + l * 12;
    gemm(xbf, nullptr, Dd, wqkv[l], Dd, P(e + 1), qkvb, nullptr, BL, 3 * Dd, Dd, false);
    k_attn<<<Bb * Hh, 256, 0, stream>>>(qkvb, mask, attnbf);
    gemm(attnbf, nullptr, Dd, wout[l], Dd, P(e + 3), projb, nullptr, BL, Dd, Dd, false);
    k_add_ln<1, 0><<<BL, 256, 0, stream>>>(xbuf, projb, P(e + 4), P(e + 5), xbuf, xbf, Dd);
    gemm(xbf, nullptr, Dd, wlin1[l], Dd, P(e + 7), ffb, ffbf, BL, FFf, Dd, true);
    gemm(ffbf, nullptr, FFf, wlin2[l], FFf, P(e + 9), ff2b, nullptr, BL, Dd, FFf, false);
    k_add_ln<1, 0><<<BL, 256, 0, stream>>>(xbuf, ff2b, P(e + 10), P(e + 11), xbuf, xbf, Dd);
  }

  // ===== composition discovery =====
  k_mean_rows<<<Bb, 256, 0, stream>>>(xbuf, pooled, pooledbf, Ll, Dd);
  gemm(pooledbf, nullptr, Dd, wog1, Dd, P(40), ogh, oghbf, Bb, 2 * Dd, Dd, true);
  gemm(oghbf, nullptr, 2 * Dd, wog2, 2 * Dd, P(42), cwb, nullptr, Bb, 64, 2 * Dd, false);
  k_softmax8<<<1, 128, 0, stream>>>(cwb);

  // ===== executor =====
  k_combined<<<Bb * Nn, 256, 0, stream>>>(components, cwb, combinedb, combbf);
  k_mean_rows<<<Bb * NCc, 256, 0, stream>>>(combinedb, selmean, nullptr, Nn, Dd);
  k_sel<<<1, 512, 0, stream>>>(selmean, pooled, P(43), P(44), selb);

  for (int k = 0; k < 4; ++k) {
    int o = 45 + k * 8;
    gemm(combbf, nullptr, Dd, wop1[k], Dd, P(o + 1), oph, nullptr, TOT, 2 * Dd, Dd, false);
    k_add_ln<0, 1><<<TOT, 256, 0, stream>>>(oph, nullptr, P(o + 2), P(o + 3), oph, ophbf, 2 * Dd);
    gemm(ophbf, nullptr, 2 * Dd, wop2[k], 2 * Dd, P(o + 5), opo, nullptr, TOT, Dd, 2 * Dd, false);
    if (k == 0) k_op_ln_axpy<1><<<TOT, 256, 0, stream>>>(opo, P(o + 6), P(o + 7), selb, k, operatedb);
    else        k_op_ln_axpy<0><<<TOT, 256, 0, stream>>>(opo, P(o + 6), P(o + 7), selb, k, operatedb);
  }

  // ===== recursive composer =====
  const int eg = (BN * Dd) / 256;
  k_gather<<<eg, 256, 0, stream>>>(operatedb, curb, curbf, 0);
  for (int d = 0; d < 3; ++d) {
    k_mean_rows<<<Bb, 256, 0, stream>>>(curb, cmean, nullptr, Nn, Dd);
    k_gate<<<1, 256, 0, stream>>>(cmean, P(77), P(78), gate, d);
    int c = 79 + d * 4;
    const __bf16* srcbf = curbf;
    float* dst = nb0;  __bf16* dstbf = nb0bf;
    for (int i = 1; i < NCc; ++i) {
      k_gather<<<eg, 256, 0, stream>>>(operatedb, nullptr, fbufbf, i);
      // cat([src, f]) @ W.T + b  (W is [768, 1536] row-major)
      gemm(srcbf, fbufbf, Dd, wcomp[d], 2 * Dd, P(c + 1), dst, nullptr, BN, Dd, Dd, false);
      k_add_ln<0, 1><<<BN, 256, 0, stream>>>(dst, nullptr, P(c + 2), P(c + 3), dst, dstbf, Dd);
      srcbf = dstbf;
      if (dst == nb0) { dst = nb1; dstbf = nb1bf; }
      else            { dst = nb0; dstbf = nb0bf; }
    }
    const float* newf = (srcbf == nb0bf) ? nb0 : nb1;
    k_blend<<<eg, 256, 0, stream>>>(curb, newf, gate, curbf);
  }

  hipMemcpyAsync(d_out, curb, (size_t)BN * Dd * sizeof(float),
                 hipMemcpyDeviceToDevice, stream);
}
